// DotAtten_62002147885096
// MI455X (gfx1250) — matrix operand order repaired and checked by reading, not compiled, not run on hardware
//
#include <hip/hip_runtime.h>
#include <hip/hip_bf16.h>
#include <math.h>

// Problem constants (match reference)
constexpr int Bb = 2;
constexpr int Ss = 512;
constexpr int E2 = 512;   // 2*encode
constexpr int Ee = 256;   // encode

constexpr int NG_COLS = 32;        // columns staged per N-group
constexpr int KPAD    = 520;       // LDS column stride in halves (bank-skewed)

typedef __attribute__((ext_vector_type(16))) _Float16 v16h;
typedef __attribute__((ext_vector_type(8)))  _Float16 v8h;
typedef __attribute__((ext_vector_type(8)))  float    v8f;

// gfx1250 has V_TANH_F32 (TRANS op). Use it; otherwise a branch-free fallback.
__device__ __forceinline__ float fast_tanhf(float x) {
#if __has_builtin(__builtin_amdgcn_tanhf)
    return __builtin_amdgcn_tanhf(x);
#else
    const float t = __expf(-2.0f * fabsf(x));
    const float y = (1.0f - t) * __builtin_amdgcn_rcpf(1.0f + t);
    return (x < 0.0f) ? -y : y;
#endif
}

// ---------- prep: fp32 -> fp16 conversions ----------
__global__ void cvt_query_f16(const float* __restrict__ q,
                              _Float16* __restrict__ qh, int n) {
    int i = blockIdx.x * blockDim.x + threadIdx.x;
    if (i < n) qh[i] = (_Float16)q[i];
}

// Wd is [E2, Ee] row-major; produce whT [Ee, E2] (column-major Wd) in fp16
__global__ void cvt_wdT_f16(const float* __restrict__ Wd,
                            _Float16* __restrict__ whT) {
    int i = blockIdx.x * blockDim.x + threadIdx.x;  // over Ee*E2
    if (i < Ee * E2) {
        int d = i / E2;
        int h = i - d * E2;
        whT[i] = (_Float16)Wd[h * Ee + d];
    }
}

// ---------- main: one workgroup per (b, s) ----------
__launch_bounds__(256)
__global__ void dot_atten_kernel(const _Float16* __restrict__ qh,    // [B,S,E2] f16
                                 const _Float16* __restrict__ whT,   // [Ee,E2] f16
                                 const float* __restrict__ value,    // [B,S,E2] f32
                                 const float* __restrict__ vd,       // [Ee]
                                 float* __restrict__ ctx_out,        // [B,S,E2]
                                 float* __restrict__ attn_out) {     // [B,S,S]
    __shared__ _Float16 bsh[NG_COLS * KPAD];  // staged Wd^T columns (~32.5 KB)
    __shared__ _Float16 scale_sh[E2];         // q[b,s,:] in f16
    __shared__ float    sjt_sh[Ss];           // logits, then softmax weights
    __shared__ float    red_sh[256];

    const int s    = blockIdx.x;
    const int b    = blockIdx.y;
    const int tid  = threadIdx.x;
    const int lane = tid & 31;
    const int wave = tid >> 5;
    const int half = lane >> 4;     // 0: lanes 0-15, 1: lanes 16-31
    const int li   = lane & 15;

    const _Float16* qbase = qh + (size_t)b * Ss * E2;

    // stage scale row + zero logits
    {
        const _Float16* srow = qbase + (size_t)s * E2;
        for (int i = tid; i < E2; i += 256) scale_sh[i] = srow[i];
        for (int i = tid; i < Ss; i += 256) sjt_sh[i] = 0.0f;
    }
    __syncthreads();

    // 8 N-groups of 32 columns; per group: stage B in LDS, then 16 M-strips.
    #pragma unroll 1
    for (int ng = 0; ng < Ee / NG_COLS; ++ng) {
        const int n0 = ng * NG_COLS;

        // ---- cooperative stage of whT[n0..n0+31][0..511] into LDS ----
        #pragma unroll 1
        for (int idx = tid; idx < NG_COLS * (E2 / 8); idx += 256) {
            const int c  = idx >> 6;          // 0..31
            const int kc = (idx & 63) << 3;   // 0,8,...,504
            *(v8h*)&bsh[c * KPAD + kc] =
                *(const v8h*)(whT + (size_t)(n0 + c) * E2 + kc);
        }
        __syncthreads();

        const int colN0 = n0 + li;
        const float v0 = vd[colN0];
        const float v1 = vd[colN0 + 16];

        // ---- 16 double-M-tile tasks (32 rows each) across 8 waves ----
        #pragma unroll 1
        for (int mt2 = wave; mt2 < 16; mt2 += 8) {
            const int m0 = mt2 << 5;          // 32-row strip
            v8f acc00 = {}, acc01 = {}, acc10 = {}, acc11 = {};

            const _Float16* aptr0 = qbase + (size_t)(m0 + li) * E2;
            const _Float16* aptr1 = qbase + (size_t)(m0 + 16 + li) * E2;

            #pragma unroll 1
            for (int kk = 0; kk < 16; ++kk) {
                const int k0 = kk << 5;

                // shared scale halves for both A fragments
                const int ka = k0 + (half << 3);
                const v8h slo = *(const v8h*)(&scale_sh[ka]);
                const v8h shi = *(const v8h*)(&scale_sh[ka + 16]);

                // A fragments (interleaved 16x32 f16 layout)
                v8h a0lo = *(const v8h*)(aptr0 + ka);
                v8h a0hi = *(const v8h*)(aptr0 + ka + 16);
                v8h a1lo = *(const v8h*)(aptr1 + ka);
                v8h a1hi = *(const v8h*)(aptr1 + ka + 16);
                a0lo *= slo;  a0hi *= shi;
                a1lo *= slo;  a1hi *= shi;
                v16h afrag0 = __builtin_shufflevector(a0lo, a0hi,
                    0,1,2,3,4,5,6,7,8,9,10,11,12,13,14,15);
                v16h afrag1 = __builtin_shufflevector(a1lo, a1hi,
                    0,1,2,3,4,5,6,7,8,9,10,11,12,13,14,15);

                // B fragments from LDS (K-major, padded stride)
                const int kb = k0 + (half << 3);
                const _Float16* bp0 = &bsh[(li)      * KPAD + kb];
                const _Float16* bp1 = &bsh[(li + 16) * KPAD + kb];
                v16h b0 = __builtin_shufflevector(*(const v8h*)bp0,
                                                  *(const v8h*)(bp0 + 16),
                    0,1,2,3,4,5,6,7,8,9,10,11,12,13,14,15);
                v16h b1 = __builtin_shufflevector(*(const v8h*)bp1,
                                                  *(const v8h*)(bp1 + 16),
                    0,1,2,3,4,5,6,7,8,9,10,11,12,13,14,15);

                acc00 = __builtin_amdgcn_wmma_f32_16x16x32_f16(
                    false, afrag0, false, b0, (short)0, acc00, false, false);
                acc01 = __builtin_amdgcn_wmma_f32_16x16x32_f16(
                    false, afrag0, false, b1, (short)0, acc01, false, false);
                acc10 = __builtin_amdgcn_wmma_f32_16x16x32_f16(
                    false, afrag1, false, b0, (short)0, acc10, false, false);
                acc11 = __builtin_amdgcn_wmma_f32_16x16x32_f16(
                    false, afrag1, false, b1, (short)0, acc11, false, false);
            }

            // Epilogue: tanh, dot with vd, reduce across 16-lane half, LDS atomic.
            #pragma unroll
            for (int r = 0; r < 8; ++r) {
                float p0 = fast_tanhf(acc00[r]) * v0 + fast_tanhf(acc01[r]) * v1;
                float p1 = fast_tanhf(acc10[r]) * v0 + fast_tanhf(acc11[r]) * v1;
                p0 += __shfl_xor(p0, 1, 32);
                p1 += __shfl_xor(p1, 1, 32);
                p0 += __shfl_xor(p0, 2, 32);
                p1 += __shfl_xor(p1, 2, 32);
                p0 += __shfl_xor(p0, 4, 32);
                p1 += __shfl_xor(p1, 4, 32);
                p0 += __shfl_xor(p0, 8, 32);
                p1 += __shfl_xor(p1, 8, 32);
                if (li == 0) {
                    atomicAdd(&sjt_sh[m0 + r + (half << 3)], p0);
                    atomicAdd(&sjt_sh[m0 + 16 + r + (half << 3)], p1);
                }
            }
        }
        __syncthreads();   // all tasks done before restaging B
    }

    // ---- softmax over t (512 logits, 256 threads) ----
    float x0 = sjt_sh[tid];
    float x1 = sjt_sh[tid + 256];
    red_sh[tid] = fmaxf(x0, x1);
    __syncthreads();
    for (int off = 128; off > 0; off >>= 1) {
        if (tid < off) red_sh[tid] = fmaxf(red_sh[tid], red_sh[tid + off]);
        __syncthreads();
    }
    const float mx = red_sh[0];
    __syncthreads();
    const float e0 = __expf(x0 - mx);
    const float e1 = __expf(x1 - mx);
    red_sh[tid] = e0 + e1;
    __syncthreads();
    for (int off = 128; off > 0; off >>= 1) {
        if (tid < off) red_sh[tid] += red_sh[tid + off];
        __syncthreads();
    }
    const float inv = 1.0f / red_sh[0];
    __syncthreads();
    const float a0 = e0 * inv;
    const float a1 = e1 * inv;
    sjt_sh[tid] = a0;
    sjt_sh[tid + 256] = a1;
    float* arow = attn_out + ((size_t)b * Ss + s) * Ss;
    arow[tid] = a0;
    arow[tid + 256] = a1;
    __syncthreads();

    // ---- context[b,s,e] = sum_t w[t] * value[b,t,e]  (coalesced along e) ----
    const float* vb = value + (size_t)b * Ss * E2;
    float c0 = 0.0f, c1 = 0.0f;
    #pragma unroll 4
    for (int t = 0; t < Ss; ++t) {
        const float w = sjt_sh[t];
        c0 += w * vb[(size_t)t * E2 + tid];
        c1 += w * vb[(size_t)t * E2 + tid + 256];
    }
    float* crow = ctx_out + ((size_t)b * Ss + s) * E2;
    crow[tid] = c0;
    crow[tid + 256] = c1;
}

extern "C" void kernel_launch(void* const* d_in, const int* in_sizes, int n_in,
                              void* d_out, int out_size, void* d_ws, size_t ws_size,
                              hipStream_t stream) {
    const float* query = (const float*)d_in[0];   // [B,S,E2]
    const float* value = (const float*)d_in[1];   // [B,S,E2]
    const float* Wd    = (const float*)d_in[2];   // [E2,Ee]
    const float* vd    = (const float*)d_in[3];   // [Ee]

    float* ctx_out  = (float*)d_out;                    // [B,S,E2]
    float* attn_out = ctx_out + (size_t)Bb * Ss * E2;   // [B,S,S]

    _Float16* qh  = (_Float16*)d_ws;                    // [B,S,E2] f16 (1 MB)
    _Float16* whT = qh + (size_t)Bb * Ss * E2;          // [Ee,E2] f16 (256 KB)

    const int nq = Bb * Ss * E2;
    cvt_query_f16<<<(nq + 255) / 256, 256, 0, stream>>>(query, qh, nq);
    const int nw = Ee * E2;
    cvt_wdT_f16<<<(nw + 255) / 256, 256, 0, stream>>>(Wd, whT);

    dim3 grid(Ss, Bb);   // one workgroup per (s, b)
    dot_atten_kernel<<<grid, 256, 0, stream>>>(qh, whT, value, vd, ctx_out, attn_out);
}
